// GAT_16587163697177
// MI455X (gfx1250) — matrix operand order repaired and checked by reading, not compiled, not run on hardware
//
#include <hip/hip_runtime.h>

typedef __attribute__((ext_vector_type(16))) _Float16 v16h;
typedef __attribute__((ext_vector_type(8)))  float    v8f;

#define N_NODES 50000
#define E_EDGES 800000
#define SLOPE   0.2f

// ---------------------------------------------------------------------------
// Weight convert + transpose: W[K][Nout] f32 (row-major)  ->  Wt[Nout][K] f16
// ---------------------------------------------------------------------------
__global__ void k_wt(const float* __restrict__ W, _Float16* __restrict__ Wt,
                     int K, int Nout) {
  int t = blockIdx.x * blockDim.x + threadIdx.x;
  if (t >= K * Nout) return;
  int k = t / Nout, n = t - k * Nout;
  Wt[n * K + k] = (_Float16)W[t];
}

// ---------------------------------------------------------------------------
// Fused dual WMMA GEMM:
//   YL[M][Nout] = X*Wl + bl ;  YR[M][Nout] = X*Wr + br
// One wave32 per 16x16 tile position; the A fragment (activations, f32->f16)
// is loaded/converted ONCE and feeds two v_wmma_f32_16x16x32_f16 per K-step.
// ---------------------------------------------------------------------------
__global__ void k_gemm2(const float* __restrict__ X,
                        const _Float16* __restrict__ WtL,
                        const _Float16* __restrict__ WtR,
                        const float* __restrict__ bl, const float* __restrict__ br,
                        float* __restrict__ YL, float* __restrict__ YR,
                        int M, int K, int Nout) {
  int wave = (int)((blockIdx.x * blockDim.x + threadIdx.x) >> 5);
  int lane = threadIdx.x & 31;
  int ntn  = Nout >> 4;
  int total = (M >> 4) * ntn;
  if (wave >= total) return;              // wave-uniform: EXEC stays all-ones
  int mt = wave / ntn, nt = wave - mt * ntn;
  int l15 = lane & 15;
  int grp = lane >> 4;                    // 0 or 1 (half-wave K group)
  int row = (mt << 4) + l15;              // A row for this lane
  int col = (nt << 4) + l15;              // B/D column for this lane

  v8f cl = {}, cr = {};
  for (int ks = 0; ks < K; ks += 32) {
    v16h a, b0, b1;
    // A fragment (16x32 f16): grp0 holds K=[ks..+7],[ks+16..+23]; grp1 offset +8
    const float* xp = X + (size_t)row * K + ks + grp * 8;
#pragma unroll
    for (int e = 0; e < 8; ++e) {
      a[e]     = (_Float16)xp[e];
      a[e + 8] = (_Float16)xp[16 + e];
    }
    // B fragments (32x16 f16): grp0 holds K=[ks..+7],[ks+16..+23]; grp1 offset +8
    const _Float16* wp0 = WtL + (size_t)col * K + ks + grp * 8;
    const _Float16* wp1 = WtR + (size_t)col * K + ks + grp * 8;
#pragma unroll
    for (int e = 0; e < 8; ++e) {
      b0[e] = wp0[e]; b0[e + 8] = wp0[16 + e];
      b1[e] = wp1[e]; b1[e + 8] = wp1[16 + e];
    }

    cl = __builtin_amdgcn_wmma_f32_16x16x32_f16(false, a, false, b0,
                                                (short)0, cl, false, false);
    cr = __builtin_amdgcn_wmma_f32_16x16x32_f16(false, a, false, b1,
                                                (short)0, cr, false, false);
  }

  float bvl = bl[col], bvr = br[col];
#pragma unroll
  for (int v = 0; v < 8; ++v) {
    int mrow = (mt << 4) + v + grp * 8;   // C/D layout: VGPR v -> row v (+8 hi half)
    YL[(size_t)mrow * Nout + col] = cl[v] + bvl;
    YR[(size_t)mrow * Nout + col] = cr[v] + bvr;
  }
}

// ---------------------------------------------------------------------------
// Init: softmax-max to -inf, denom/accumulator to zero
// ---------------------------------------------------------------------------
__global__ void k_init(float* __restrict__ m, float* __restrict__ denom,
                       float* __restrict__ acc, int nm, int nacc) {
  int t = blockIdx.x * blockDim.x + threadIdx.x;
  if (t < nm) { m[t] = -3.0e38f; denom[t] = 0.0f; }
  if (t < nacc) acc[t] = 0.0f;
}

__device__ inline void atomicMaxF(float* addr, float val) {
  unsigned int* ua = (unsigned int*)addr;
  unsigned int old = *ua;
  while (__uint_as_float(old) < val) {
    unsigned int assumed = old;
    old = atomicCAS(ua, assumed, __float_as_uint(val));
    if (old == assumed) break;
  }
}

// ---------------------------------------------------------------------------
// Edge pass 1: logits[e,h] = att_h . leakyrelu(xr[dst] + xl[src]); seg-max
// One wave per edge; lanes stride C (coalesced gathers), shfl reduction.
// ---------------------------------------------------------------------------
template <int H, int C>
__global__ void k_logits(const float* __restrict__ xl, const float* __restrict__ xr,
                         const float* __restrict__ att, const int* __restrict__ src,
                         const int* __restrict__ dst, float* __restrict__ logits,
                         float* __restrict__ m) {
  int w = (int)((blockIdx.x * blockDim.x + threadIdx.x) >> 5);
  int lane = threadIdx.x & 31;
  if (w >= E_EDGES) return;
  int s = src[w], d = dst[w];
  const float* pl = xl + (size_t)s * H * C;
  const float* pr = xr + (size_t)d * H * C;
#pragma unroll
  for (int h = 0; h < H; ++h) {
    float sum = 0.0f;
#pragma unroll
    for (int i = lane; i < C; i += 32) {
      float v = pr[h * C + i] + pl[h * C + i];
      v = v > 0.0f ? v : SLOPE * v;
      sum += v * att[h * C + i];
    }
#pragma unroll
    for (int off = 16; off; off >>= 1) sum += __shfl_xor(sum, off, 32);
    if (lane == 0) {
      logits[(size_t)w * H + h] = sum;
      atomicMaxF(&m[d * H + h], sum);
    }
  }
}

// ---------------------------------------------------------------------------
// Edge pass 2: a = exp(logit - max[dst]); denom[dst] += a (in place on logits)
// ---------------------------------------------------------------------------
template <int H>
__global__ void k_expsum(const int* __restrict__ dst, float* __restrict__ logits,
                         const float* __restrict__ m, float* __restrict__ denom) {
  int t = blockIdx.x * blockDim.x + threadIdx.x;
  if (t >= E_EDGES * H) return;
  int e = t / H, h = t - e * H;
  int d = dst[e];
  float a = __expf(logits[t] - m[d * H + h]);
  logits[t] = a;
  atomicAdd(&denom[d * H + h], a);
}

// ---------------------------------------------------------------------------
// Edge pass 3: acc[dst] += (a / denom[dst]) * xl[src]   (coalesced scatter-add)
// ---------------------------------------------------------------------------
template <int H, int C>
__global__ void k_aggr(const float* __restrict__ xl, const float* __restrict__ alpha,
                       const float* __restrict__ denom, const int* __restrict__ src,
                       const int* __restrict__ dst, float* __restrict__ acc) {
  int w = (int)((blockIdx.x * blockDim.x + threadIdx.x) >> 5);
  int lane = threadIdx.x & 31;
  if (w >= E_EDGES) return;
  int s = src[w], d = dst[w];
#pragma unroll
  for (int h = 0; h < H; ++h) {
    float al = alpha[(size_t)w * H + h] / (denom[d * H + h] + 1e-16f);
#pragma unroll
    for (int i = lane; i < C; i += 32)
      atomicAdd(&acc[(size_t)d * H * C + h * C + i],
                al * xl[(size_t)s * H * C + h * C + i]);
  }
}

// ---------------------------------------------------------------------------
// Finalize: out = acc + bias, optional ReLU
// ---------------------------------------------------------------------------
__global__ void k_final(const float* __restrict__ acc, const float* __restrict__ bias,
                        float* __restrict__ out, int n, int HC, int relu) {
  int t = blockIdx.x * blockDim.x + threadIdx.x;
  if (t >= n) return;
  float v = acc[t] + bias[t % HC];
  if (relu && v < 0.0f) v = 0.0f;
  out[t] = v;
}

// ---------------------------------------------------------------------------
// Host-side layer driver
// ---------------------------------------------------------------------------
struct Scratch {
  float *xl, *xr, *acc, *logits, *m, *denom;
  _Float16 *wlt, *wrt;
};

static inline int ceil_div(long long a, long long b) { return (int)((a + b - 1) / b); }

static void run_layer(const float* xin, int K, int H, int C,
                      const float* Wl, const float* bl,
                      const float* Wr, const float* br,
                      const float* att, const float* bias,
                      const int* src, const int* dst,
                      float* outbuf, int relu, const Scratch& S,
                      hipStream_t stream) {
  const int HC = H * C;
  const int M = N_NODES;

  // Weight convert/transpose
  k_wt<<<ceil_div((long long)K * HC, 256), 256, 0, stream>>>(Wl, S.wlt, K, HC);
  k_wt<<<ceil_div((long long)K * HC, 256), 256, 0, stream>>>(Wr, S.wrt, K, HC);

  // Node transforms (fused): XL = X*Wl + bl and XR = X*Wr + br, WMMA
  int tiles = (M >> 4) * (HC >> 4);
  int gblocks = ceil_div(tiles, 8);       // 8 waves (tiles) per 256-thread block
  k_gemm2<<<gblocks, 256, 0, stream>>>(xin, S.wlt, S.wrt, bl, br,
                                       S.xl, S.xr, M, K, HC);

  // Init softmax state + accumulator
  int nacc = M * HC;
  k_init<<<ceil_div(nacc, 256), 256, 0, stream>>>(S.m, S.denom, S.acc, M * H, nacc);

  // Edge passes
  int eblocks = ceil_div(E_EDGES, 8);     // one wave per edge, 8 waves/block
  if (H == 1 && C == 64) {
    k_logits<1, 64><<<eblocks, 256, 0, stream>>>(S.xl, S.xr, att, src, dst, S.logits, S.m);
    k_expsum<1><<<ceil_div(E_EDGES, 256), 256, 0, stream>>>(dst, S.logits, S.m, S.denom);
    k_aggr<1, 64><<<eblocks, 256, 0, stream>>>(S.xl, S.logits, S.denom, src, dst, S.acc);
  } else if (H == 2 && C == 64) {
    k_logits<2, 64><<<eblocks, 256, 0, stream>>>(S.xl, S.xr, att, src, dst, S.logits, S.m);
    k_expsum<2><<<ceil_div(2LL * E_EDGES, 256), 256, 0, stream>>>(dst, S.logits, S.m, S.denom);
    k_aggr<2, 64><<<eblocks, 256, 0, stream>>>(S.xl, S.logits, S.denom, src, dst, S.acc);
  } else { // H == 1, C == 128
    k_logits<1, 128><<<eblocks, 256, 0, stream>>>(S.xl, S.xr, att, src, dst, S.logits, S.m);
    k_expsum<1><<<ceil_div(E_EDGES, 256), 256, 0, stream>>>(dst, S.logits, S.m, S.denom);
    k_aggr<1, 128><<<eblocks, 256, 0, stream>>>(S.xl, S.logits, S.denom, src, dst, S.acc);
  }

  // Bias (+ReLU)
  k_final<<<ceil_div(nacc, 256), 256, 0, stream>>>(S.acc, bias, outbuf, nacc, HC, relu);
}

extern "C" void kernel_launch(void* const* d_in, const int* in_sizes, int n_in,
                              void* d_out, int out_size, void* d_ws, size_t ws_size,
                              hipStream_t stream) {
  // Inputs in setup_inputs() dict order:
  // x, edge_index, then per layer (l1,l2,l3): Wl, bl, Wr, br, att, bias
  const float* x   = (const float*)d_in[0];
  const int*   ei  = (const int*)d_in[1];
  const int*   src = ei;
  const int*   dst = ei + E_EDGES;

  const float* p[18];
  for (int i = 0; i < 18; ++i) p[i] = (const float*)d_in[2 + i];
  // layer l: p[6l+0]=Wl p[6l+1]=bl p[6l+2]=Wr p[6l+3]=br p[6l+4]=att p[6l+5]=bias

  // Workspace carve-up (256B-aligned slices)
  char* w = (char*)d_ws;
  auto alloc = [&](size_t bytes) {
    char* q = w;
    w += (bytes + 255) & ~(size_t)255;
    return q;
  };
  Scratch S;
  S.xl     = (float*)alloc((size_t)N_NODES * 128 * 4);
  S.xr     = (float*)alloc((size_t)N_NODES * 128 * 4);
  S.acc    = (float*)alloc((size_t)N_NODES * 128 * 4);
  float* h1 = (float*)alloc((size_t)N_NODES * 64 * 4);
  float* h2 = (float*)alloc((size_t)N_NODES * 128 * 4);
  S.logits = (float*)alloc((size_t)E_EDGES * 2 * 4);
  S.m      = (float*)alloc((size_t)N_NODES * 2 * 4);
  S.denom  = (float*)alloc((size_t)N_NODES * 2 * 4);
  S.wlt    = (_Float16*)alloc((size_t)128 * 128 * 2);
  S.wrt    = (_Float16*)alloc((size_t)128 * 128 * 2);
  (void)ws_size; (void)in_sizes; (void)n_in; (void)out_size;

  // Layer 1: fin=128, H=1, C=64, ReLU -> h1
  run_layer(x,  128, 1, 64,  p[0],  p[1],  p[2],  p[3],  p[4],  p[5],
            src, dst, h1, 1, S, stream);
  // Layer 2: fin=64, H=2, C=64, ReLU -> h2
  run_layer(h1, 64,  2, 64,  p[6],  p[7],  p[8],  p[9],  p[10], p[11],
            src, dst, h2, 1, S, stream);
  // Layer 3: fin=128, H=1, C=128 -> d_out
  run_layer(h2, 128, 1, 128, p[12], p[13], p[14], p[15], p[16], p[17],
            src, dst, (float*)d_out, 0, S, stream);
}
